// SelfAttnMatch_82076825026825
// MI455X (gfx1250) — matrix operand order repaired and checked by reading, not compiled, not run on hardware
//
#include <hip/hip_runtime.h>

#define B_  32
#define L_  1024
#define D_  1024
#define SST 1028   // scores row stride in floats (pad -> bank-conflict-free b128 reads)
#define KST 1032   // stage row stride in halves  (pad -> bank-conflict-free b128 reads)
#define QT  32     // query rows per workgroup

typedef _Float16 v16h __attribute__((ext_vector_type(16)));
typedef _Float16 v8h  __attribute__((ext_vector_type(8)));
typedef _Float16 v4h  __attribute__((ext_vector_type(4)));
typedef float    v8f  __attribute__((ext_vector_type(8)));

union V16u { v16h v; v8h h2[2]; };

// ---- fragment builders -------------------------------------------------------
// A-fragment (16x32 f16): lane (m=l%16, h=l/16) holds K = h*16+{0..15} contiguous.

// from fp32 storage: p -> (row, k0 + h*16)
__device__ __forceinline__ v16h load_a_cvt(const float* p) {
  float4 f0 = *(const float4*)(p);
  float4 f1 = *(const float4*)(p + 4);
  float4 f2 = *(const float4*)(p + 8);
  float4 f3 = *(const float4*)(p + 12);
  v16h a;
  a[0]=(_Float16)f0.x;  a[1]=(_Float16)f0.y;  a[2]=(_Float16)f0.z;  a[3]=(_Float16)f0.w;
  a[4]=(_Float16)f1.x;  a[5]=(_Float16)f1.y;  a[6]=(_Float16)f1.z;  a[7]=(_Float16)f1.w;
  a[8]=(_Float16)f2.x;  a[9]=(_Float16)f2.y;  a[10]=(_Float16)f2.z; a[11]=(_Float16)f2.w;
  a[12]=(_Float16)f3.x; a[13]=(_Float16)f3.y; a[14]=(_Float16)f3.z; a[15]=(_Float16)f3.w;
  return a;
}

// from f16 storage: p -> (row, k0 + h*16); chunks at +0 and +8 halves
__device__ __forceinline__ v16h load_a16(const _Float16* p) {
  V16u u;
  u.h2[0] = *(const v8h*)(p);
  u.h2[1] = *(const v8h*)(p + 8);
  return u.v;
}

// B-fragment (32x16 f16): lane (n=l%16, h=l/16) holds K = h*16+{0..15} contiguous.
__device__ __forceinline__ v16h load_b16(const _Float16* p) {
  V16u u;
  u.h2[0] = *(const v8h*)(p);
  u.h2[1] = *(const v8h*)(p + 8);
  return u.v;
}

__device__ __forceinline__ v16h load_b_cvt(const float* p) {
  float4 f0 = *(const float4*)(p);
  float4 f1 = *(const float4*)(p + 4);
  float4 f2 = *(const float4*)(p + 8);
  float4 f3 = *(const float4*)(p + 12);
  v16h a;
  a[0]=(_Float16)f0.x;  a[1]=(_Float16)f0.y;  a[2]=(_Float16)f0.z;  a[3]=(_Float16)f0.w;
  a[4]=(_Float16)f1.x;  a[5]=(_Float16)f1.y;  a[6]=(_Float16)f1.z;  a[7]=(_Float16)f1.w;
  a[8]=(_Float16)f2.x;  a[9]=(_Float16)f2.y;  a[10]=(_Float16)f2.z; a[11]=(_Float16)f2.w;
  a[12]=(_Float16)f3.x; a[13]=(_Float16)f3.y; a[14]=(_Float16)f3.z; a[15]=(_Float16)f3.w;
  return a;
}

// ---- pre-pass: x (fp32) -> xh (f16), one shot, L2-resident afterwards --------
__global__ __launch_bounds__(256)
void cvt_f16_kernel(const float* __restrict__ x, _Float16* __restrict__ xh) {
  size_t i = ((size_t)blockIdx.x * 256 + threadIdx.x) * 8;
  float4 f0 = *(const float4*)(x + i);
  float4 f1 = *(const float4*)(x + i + 4);
  v8h h;
  h[0]=(_Float16)f0.x; h[1]=(_Float16)f0.y; h[2]=(_Float16)f0.z; h[3]=(_Float16)f0.w;
  h[4]=(_Float16)f1.x; h[5]=(_Float16)f1.y; h[6]=(_Float16)f1.z; h[7]=(_Float16)f1.w;
  *(v8h*)(xh + i) = h;
}

// ---- fused attention kernel --------------------------------------------------
__global__ __launch_bounds__(128, 1)
void selfattn_kernel(const float*    __restrict__ x,
                     const _Float16* __restrict__ xh,
                     const int*      __restrict__ xmask,
                     float*          __restrict__ out,
                     int useh)
{
  extern __shared__ char smem[];
  float*    maskf = (float*)smem;                                // 1024 f32
  float*    sc    = (float*)(smem + 4096);                       // [QT][SST] f32
  _Float16* stage = (_Float16*)(smem + 4096 + QT * SST * 4);     // [64][KST] f16

  const int b    = blockIdx.y;
  const int q0   = blockIdx.x * QT;
  const int tid  = threadIdx.x;
  const int lane = tid & 31;
  const int wv   = tid >> 5;
  const int rw   = wv >> 1;      // which 16-row group this wave owns
  const int cw   = wv & 1;       // which half (d-half in P1, d-columns in P3)
  const int hh   = lane >> 4;    // lane half
  const int mr   = lane & 15;    // m (A rows) / n (B cols) index

  const float*    xb  = x  + (size_t)b * (L_ * D_);
  const _Float16* xhb = xh + (size_t)b * (L_ * D_);

  for (int j = tid; j < L_; j += 128) maskf[j] = (float)xmask[b * L_ + j];
  for (int i = tid; i < QT * SST; i += 128) sc[i] = 0.0f;
  __syncthreads();

  // ---------------- Phase 1: Q cache (16 rows x 512 d per wave, f16 in VGPRs) ----
  v16h qc[16];
  if (useh) {
    const _Float16* qrow = xhb + (size_t)(q0 + rw * 16 + mr) * D_ + cw * 512;
    #pragma unroll
    for (int kk = 0; kk < 16; ++kk)
      qc[kk] = load_a16(qrow + kk * 32 + hh * 16);
  } else {
    const float* qrow = xb + (size_t)(q0 + rw * 16 + mr) * D_ + cw * 512;
    #pragma unroll
    for (int kk = 0; kk < 16; ++kk)
      qc[kk] = load_a_cvt(qrow + kk * 32 + hh * 16);
  }

  // ---------------- Phase 1: scores = Q @ K^T, B-fragments straight from L2 ----
  // 1-stage software pipeline: fragments kk+2/kk+3 are in flight while the WMMAs
  // for kk/kk+1 issue, so the matrix pipe never waits on loadcnt==0.
  for (int kt = 0; kt < 64; ++kt) {
    v8f c0, c1;
    #pragma unroll
    for (int i = 0; i < 8; ++i) { c0[i] = 0.0f; c1[i] = 0.0f; }
    if (useh) {
      const _Float16* kb = xhb + (size_t)(kt * 16 + mr) * D_ + cw * 512 + hh * 16;
      v16h bf0 = load_b16(kb);
      v16h bf1 = load_b16(kb + 32);
      #pragma unroll
      for (int kk = 0; kk < 16; kk += 2) {
        v16h nf0 = bf0, nf1 = bf1;
        if (kk + 2 < 16) {
          nf0 = load_b16(kb + (kk + 2) * 32);
          nf1 = load_b16(kb + (kk + 3) * 32);
        }
        c0 = __builtin_amdgcn_wmma_f32_16x16x32_f16(false, qc[kk + 0], false, bf0,
                                                    (short)0, c0, false, false);
        c1 = __builtin_amdgcn_wmma_f32_16x16x32_f16(false, qc[kk + 1], false, bf1,
                                                    (short)0, c1, false, false);
        bf0 = nf0; bf1 = nf1;
      }
    } else {
      const float* kb = xb + (size_t)(kt * 16 + mr) * D_ + cw * 512 + hh * 16;
      v16h bf0 = load_b_cvt(kb);
      v16h bf1 = load_b_cvt(kb + 32);
      #pragma unroll
      for (int kk = 0; kk < 16; kk += 2) {
        v16h nf0 = bf0, nf1 = bf1;
        if (kk + 2 < 16) {
          nf0 = load_b_cvt(kb + (kk + 2) * 32);
          nf1 = load_b_cvt(kb + (kk + 3) * 32);
        }
        c0 = __builtin_amdgcn_wmma_f32_16x16x32_f16(false, qc[kk + 0], false, bf0,
                                                    (short)0, c0, false, false);
        c1 = __builtin_amdgcn_wmma_f32_16x16x32_f16(false, qc[kk + 1], false, bf1,
                                                    (short)0, c1, false, false);
        bf0 = nf0; bf1 = nf1;
      }
    }
    // combine the two d-half partials (cw=0 and cw=1) into the score buffer
    #pragma unroll
    for (int r = 0; r < 8; ++r)
      atomicAdd(&sc[(rw * 16 + r + 8 * hh) * SST + kt * 16 + mr], c0[r] + c1[r]);
  }
  __syncthreads();

  // ---------------- Phase 2: masked softmax (equiv. to ref's a/sum(a) form) ----
  {
    int row = wv * 8;   // 8 rows per wave
    for (int rr = 0; rr < 8; ++rr, ++row) {
      const int grow = q0 + row;
      float sv[32];
      float mx = -3.0e38f;
      #pragma unroll
      for (int jj = 0; jj < 32; ++jj) {
        int j = jj * 32 + lane;
        float s = sc[row * SST + j];
        if (j == grow) s = 0.0f;               // zeroed diagonal
        sv[jj] = s;
        if (maskf[j] > 0.5f && s > mx) mx = s; // max over unmasked cols only
      }
      #pragma unroll
      for (int off = 16; off > 0; off >>= 1)
        mx = fmaxf(mx, __shfl_xor(mx, off, 32));
      float sum = 0.0f;
      if (mx > -1.0e38f) {
        #pragma unroll
        for (int jj = 0; jj < 32; ++jj) {
          int j = jj * 32 + lane;
          float p = (maskf[j] > 0.5f) ? __expf(sv[jj] - mx) : 0.0f;
          sv[jj] = p; sum += p;
        }
      } else {
        #pragma unroll
        for (int jj = 0; jj < 32; ++jj) sv[jj] = 0.0f;  // fully masked row -> zeros
      }
      #pragma unroll
      for (int off = 16; off > 0; off >>= 1)
        sum += __shfl_xor(sum, off, 32);
      float inv = 1.0f / (sum + 1e-13f);
      #pragma unroll
      for (int jj = 0; jj < 32; ++jj)
        sc[row * SST + jj * 32 + lane] = sv[jj] * inv;   // alpha, in place (fp32)
    }
  }

  // ---------------- Phase 3: out = alpha @ V (V^T tile staged in LDS) ----
  for (int dg = 0; dg < 16; ++dg) {
    __syncthreads();
    if (useh) {
      for (int i2 = 0; i2 < 128; ++i2) {
        int f4  = i2 * 128 + tid;
        int key = f4 >> 4;
        int dl  = (f4 & 15) * 4;
        v4h v = *(const v4h*)(xhb + (size_t)key * D_ + dg * 64 + dl);
        stage[(dl + 0) * KST + key] = v[0];
        stage[(dl + 1) * KST + key] = v[1];
        stage[(dl + 2) * KST + key] = v[2];
        stage[(dl + 3) * KST + key] = v[3];
      }
    } else {
      for (int i2 = 0; i2 < 128; ++i2) {
        int f4  = i2 * 128 + tid;
        int key = f4 >> 4;
        int dl  = (f4 & 15) * 4;
        float4 v = *(const float4*)(xb + (size_t)key * D_ + dg * 64 + dl);
        stage[(dl + 0) * KST + key] = (_Float16)v.x;
        stage[(dl + 1) * KST + key] = (_Float16)v.y;
        stage[(dl + 2) * KST + key] = (_Float16)v.z;
        stage[(dl + 3) * KST + key] = (_Float16)v.w;
      }
    }
    __syncthreads();
    v8f a0, a1;
    #pragma unroll
    for (int i = 0; i < 8; ++i) { a0[i] = 0.0f; a1[i] = 0.0f; }
    #pragma unroll 4
    for (int kc = 0; kc < 32; ++kc) {
      v16h af = load_a_cvt(&sc[(rw * 16 + mr) * SST + kc * 32 + hh * 16]);
      v16h b0 = load_b16(stage + (cw * 32 + mr)      * KST + kc * 32 + hh * 16);
      v16h b1 = load_b16(stage + (cw * 32 + 16 + mr) * KST + kc * 32 + hh * 16);
      a0 = __builtin_amdgcn_wmma_f32_16x16x32_f16(false, af, false, b0,
                                                  (short)0, a0, false, false);
      a1 = __builtin_amdgcn_wmma_f32_16x16x32_f16(false, af, false, b1,
                                                  (short)0, a1, false, false);
    }
    const int col0 = dg * 64 + cw * 32 + mr;
    #pragma unroll
    for (int r = 0; r < 8; ++r) {
      size_t o = ((size_t)b * L_ + (q0 + rw * 16 + r + 8 * hh)) * D_ + col0;
      out[o]      = a0[r];
      out[o + 16] = a1[r];
    }
  }
}

extern "C" void kernel_launch(void* const* d_in, const int* in_sizes, int n_in,
                              void* d_out, int out_size, void* d_ws, size_t ws_size,
                              hipStream_t stream) {
  const float* x     = (const float*)d_in[0];
  const int*   xmask = (const int*)d_in[1];
  float*       out   = (float*)d_out;

  const size_t nelem = (size_t)B_ * L_ * D_;
  const int    useh  = (ws_size >= nelem * sizeof(_Float16)) ? 1 : 0;
  _Float16*    xh    = (_Float16*)d_ws;

  if (useh) {
    // one-shot fp32 -> f16 pre-pass; xh (64 MB) stays resident in the 192 MB L2
    cvt_f16_kernel<<<dim3((unsigned)(nelem / (256 * 8))), dim3(256), 0, stream>>>(x, xh);
  }

  const size_t smem = 4096 + (size_t)QT * SST * 4 + (size_t)64 * KST * 2; // ~261.5 KB
  (void)hipFuncSetAttribute((const void*)selfattn_kernel,
                            hipFuncAttributeMaxDynamicSharedMemorySize, (int)smem);
  dim3 grid(L_ / QT, B_);
  selfattn_kernel<<<grid, dim3(128, 1, 1), smem, stream>>>(x, xh, xmask, out, useh);
}
